// HeteroGraphHGT_17428977287425
// MI455X (gfx1250) — matrix operand order repaired and checked by reading, not compiled, not run on hardware
//
#include <hip/hip_runtime.h>
#include <hip/hip_bf16.h>
#include <math.h>

// ---------------------------------------------------------------------------
// HGT GNN forward for MI455X (gfx1250, wave32, WMMA bf16 16x16x32)
//  - weights pre-swizzled to WMMA B-fragment order (32B contiguous per lane)
//  - N = 50000 = 3125 * 16  -> no row guards anywhere
//  - row-strip outputs staged in LDS, flushed with coalesced b128 stores
// ---------------------------------------------------------------------------

#define NN  50000
#define EE  800000
#define HH  128
#define BB  512
#define NL  3
#define SCALE_F 0.17677669529663687f  /* 1/sqrt(32) */

typedef __attribute__((ext_vector_type(16))) __bf16 v16bf;
typedef __attribute__((ext_vector_type(8)))  float  v8f;

// Swizzled weight layout: idx = ((tile*(K/32) + kc)*32 + lane)*16 + j
//   element j of lane's B fragment = W[(kc*32 + (lane>>4)*16 + j)][tile*16 + (lane&15)]
// Each lane's 16 bf16 fragment is 32B contiguous & 32B aligned.

// ---------------------------------------------------------------------------
// WMMA tile: acc += A(16 x KTOT bf16, LDS row-major stride sas) x Wswz tile
// ---------------------------------------------------------------------------
template <int KTOT>
__device__ __forceinline__ v8f wmma_tile(const __bf16* sA, int sas,
                                         const __bf16* Wswz, int tileIdx, v8f acc) {
    const int lane = threadIdx.x & 31;
    const int r    = lane & 15;
    const int half = lane >> 4;
    const __bf16* wp = Wswz + ((size_t)tileIdx * (KTOT / 32) * 32 + lane) * 16;
#pragma unroll
    for (int kk = 0; kk < KTOT; kk += 32) {
        v16bf a;
        const int ka = kk + half * 16;
#pragma unroll
        for (int j = 0; j < 8; ++j) a[j]     = sA[r * sas + ka + j];
#pragma unroll
        for (int j = 0; j < 8; ++j) a[8 + j] = sA[r * sas + ka + 8 + j];
        const v16bf b = *(const v16bf*)wp;
        wp += 32 * 16;
        acc = __builtin_amdgcn_wmma_f32_16x16x32_bf16(
            false, a, false, b, (short)0, acc, false, false);
    }
    return acc;
}

// Stage 16x16 f32 tile (+bias) into LDS sOut[16][128]
__device__ __forceinline__ void stage_tile(float* sOut, v8f acc, int colBase,
                                           const float* bias) {
    const int lane = threadIdx.x & 31;
    const int c    = colBase + (lane & 15);
    const int rh   = (lane >> 4) * 8;
    const float bv = bias[c];
#pragma unroll
    for (int v = 0; v < 8; ++v) sOut[(rh + v) * 128 + c] = acc[v] + bv;
}

// Coalesced flush: 16x128 floats = 512 float4, 256 threads x 2
__device__ __forceinline__ void flush_out(const float* sOut, float* out, int row0) {
    float4* o = (float4*)out + (size_t)row0 * 32;
    const float4* s = (const float4*)sOut;
#pragma unroll
    for (int i = threadIdx.x; i < 512; i += 256) o[i] = s[i];
}

// ---------------------------------------------------------------------------
// Kernel 0: weight prep — cast to bf16 in swizzled fragment order;
// fold a_rel/m_rel into k_W/v_W (+ fused biases).
// Grid: 128 blocks x 128 threads (t = 0..16383)
// ---------------------------------------------------------------------------
__global__ __launch_bounds__(128) void prep_kernel(
    const float* lin_op_W, const float* kW, const float* qW, const float* vW,
    const float* aW, const float* kb, const float* vb,
    const float* a_rel, const float* m_rel,
    __bf16* bWop, __bf16* bWq, __bf16* bWk, __bf16* bWv, __bf16* bWa,
    float* fbk, float* fbv) {
    const int t = blockIdx.x * 128 + threadIdx.x;

    // K=128 matrices: decompose swizzled index
    {
        const int j    = t & 15;
        const int lane = (t >> 4) & 31;
        const int kc   = (t >> 9) & 3;
        const int tile = t >> 11;                    // 0..7
        const int kRow = kc * 32 + (lane >> 4) * 16 + j;
        const int col  = tile * 16 + (lane & 15);
        bWq[t] = (__bf16)qW[kRow * HH + col];
        bWa[t] = (__bf16)aW[kRow * HH + col];
        const int hg = col >> 5, e = col & 31;
        float sk = 0.f, sv = 0.f;
#pragma unroll 4
        for (int d = 0; d < 32; ++d) {
            sk += kW[kRow * HH + hg * 32 + d] * a_rel[(hg * 32 + d) * 32 + e];
            sv += vW[kRow * HH + hg * 32 + d] * m_rel[(hg * 32 + d) * 32 + e];
        }
        bWk[t] = (__bf16)sk;
        bWv[t] = (__bf16)sv;
    }
    // K=64 input-projection matrix (8192 swizzled elements)
    if (t < 8192) {
        const int j    = t & 15;
        const int lane = (t >> 4) & 31;
        const int kc   = (t >> 9) & 1;
        const int tile = t >> 10;                    // 0..7
        const int kRow = kc * 32 + (lane >> 4) * 16 + j;
        const int col  = tile * 16 + (lane & 15);
        bWop[t] = (__bf16)lin_op_W[kRow * HH + col];
    }
    // Fused biases
    if (t < 128) {
        const int hg = t >> 5, e = t & 31;
        float bk2 = 0.f, bv2 = 0.f;
        for (int d = 0; d < 32; ++d) {
            bk2 += kb[hg * 32 + d] * a_rel[(hg * 32 + d) * 32 + e];
            bv2 += vb[hg * 32 + d] * m_rel[(hg * 32 + d) * 32 + e];
        }
        fbk[t] = bk2;
        fbv[t] = bv2;
    }
}

// ---------------------------------------------------------------------------
// Kernel 1: input projection  h = x @ Wop + b  (K=64)
// ---------------------------------------------------------------------------
__global__ __launch_bounds__(256) void inproj_kernel(
    const float* x, const __bf16* W, const float* bias, float* h) {
    __shared__ __bf16 sA[16 * 72];
    __shared__ float  sOut[16 * 128];
    const int row0 = blockIdx.x * 16;
#pragma unroll
    for (int i = threadIdx.x; i < 16 * 64; i += 256) {
        const int r = i >> 6, c = i & 63;
        sA[r * 72 + c] = (__bf16)x[(size_t)(row0 + r) * 64 + c];
    }
    __syncthreads();
    const int tile = threadIdx.x >> 5;
    v8f z = {};
    v8f acc = wmma_tile<64>(sA, 72, W, tile, z);
    stage_tile(sOut, acc, tile * 16, bias);
    __syncthreads();
    flush_out(sOut, h, row0);
}

// ---------------------------------------------------------------------------
// Kernel 2: fused q / k_rel / v_rel projections (K=128, 3 GEMMs, shared A)
// ---------------------------------------------------------------------------
__global__ __launch_bounds__(256) void qkv_kernel(
    const float* h, const __bf16* Wq, const __bf16* Wk, const __bf16* Wv,
    const float* bq, const float* fbk, const float* fbv,
    float* q, float* kr, float* vr) {
    __shared__ __bf16 sA[16 * 136];
    __shared__ float  sOut[16 * 128];
    const int row0 = blockIdx.x * 16;
#pragma unroll
    for (int i = threadIdx.x; i < 16 * 128; i += 256) {
        const int r = i >> 7, c = i & 127;
        sA[r * 136 + c] = (__bf16)h[(size_t)(row0 + r) * HH + c];
    }
    __syncthreads();
    const int tile = threadIdx.x >> 5;
    v8f z = {};

    v8f aq = wmma_tile<128>(sA, 136, Wq, tile, z);
    stage_tile(sOut, aq, tile * 16, bq);
    __syncthreads();
    flush_out(sOut, q, row0);
    __syncthreads();

    v8f ak = wmma_tile<128>(sA, 136, Wk, tile, z);
    stage_tile(sOut, ak, tile * 16, fbk);
    __syncthreads();
    flush_out(sOut, kr, row0);
    __syncthreads();

    v8f av = wmma_tile<128>(sA, 136, Wv, tile, z);
    stage_tile(sOut, av, tile * 16, fbv);
    __syncthreads();
    flush_out(sOut, vr, row0);
}

// ---------------------------------------------------------------------------
// Edge phase (one thread per (edge, head))
// ---------------------------------------------------------------------------
__device__ __forceinline__ unsigned f2ord(float f) {
    const unsigned u = __float_as_uint(f);
    return (u & 0x80000000u) ? ~u : (u | 0x80000000u);
}
__device__ __forceinline__ float ord2f(unsigned v) {
    return __uint_as_float((v & 0x80000000u) ? (v & 0x7FFFFFFFu) : ~v);
}

__global__ __launch_bounds__(256) void edge_logits_kernel(
    const int* src, const int* dst, const float* q, const float* kr,
    const float* p_rel, float* elog, unsigned* nmax) {
    const int t = blockIdx.x * 256 + threadIdx.x;
    if (t >= EE * 4) return;
    const int e = t >> 2, hh = t & 3;
    const int s = src[e], d = dst[e];
    const float* qp = q  + (size_t)d * HH + hh * 32;
    const float* kp = kr + (size_t)s * HH + hh * 32;
    float acc = 0.f;
#pragma unroll
    for (int i = 0; i < 32; i += 4) {
        const float4 qa = *(const float4*)(qp + i);
        const float4 ka = *(const float4*)(kp + i);
        acc += qa.x * ka.x + qa.y * ka.y + qa.z * ka.z + qa.w * ka.w;
    }
    acc *= p_rel[hh] * SCALE_F;
    elog[t] = acc;
    atomicMax(&nmax[(size_t)d * 4 + hh], f2ord(acc));
}

__global__ __launch_bounds__(256) void edge_expsum_kernel(
    const int* dst, float* elog, const unsigned* nmax, float* nsum) {
    const int t = blockIdx.x * 256 + threadIdx.x;
    if (t >= EE * 4) return;
    const int e = t >> 2, hh = t & 3;
    const int d = dst[e];
    const float m  = ord2f(nmax[(size_t)d * 4 + hh]);
    const float ex = __expf(elog[t] - m);
    elog[t] = ex;
    atomicAdd(&nsum[(size_t)d * 4 + hh], ex);
}

__global__ __launch_bounds__(256) void edge_msg_kernel(
    const int* src, const int* dst, const float* elog, const float* nsum,
    const float* vr, float* agg) {
    const int t = blockIdx.x * 256 + threadIdx.x;
    if (t >= EE * 4) return;
    const int e = t >> 2, hh = t & 3;
    const int s = src[e], d = dst[e];
    const float alpha = elog[t] / (nsum[(size_t)d * 4 + hh] + 1e-16f);
    const float* vp = vr + (size_t)s * HH + hh * 32;
    float* op = agg + (size_t)d * HH + hh * 32;
#pragma unroll 4
    for (int i = 0; i < 32; ++i) atomicAdd(op + i, vp[i] * alpha);
}

// ---------------------------------------------------------------------------
// Kernel 3: out-projection  gelu(agg) @ a_W + a_b, fused skip + ELU + LayerNorm
// h updated in place (each block owns its 16 rows exclusively).
// ---------------------------------------------------------------------------
__global__ __launch_bounds__(256) void outproj_norm_kernel(
    const float* agg, float* h, const __bf16* Wa, const float* ba,
    const float* skip, const float* ng, const float* nb) {
    __shared__ __bf16 sA[16 * 136];
    __shared__ float  sOut[16 * 128];
    const int row0 = blockIdx.x * 16;
#pragma unroll
    for (int i = threadIdx.x; i < 16 * 128; i += 256) {
        const int r = i >> 7, c = i & 127;
        float f = agg[(size_t)(row0 + r) * HH + c];
        f = 0.5f * f * (1.f + erff(f * 0.70710678118654752f));   // exact gelu
        sA[r * 136 + c] = (__bf16)f;
    }
    __syncthreads();
    const int tile = threadIdx.x >> 5;
    v8f z = {};
    v8f acc = wmma_tile<128>(sA, 136, Wa, tile, z);
    stage_tile(sOut, acc, tile * 16, ba);
    __syncthreads();

    // Epilogue: 128 threads, 8 lanes per row, 16 cols per lane.
    if (threadIdx.x < 128) {
        const int r   = threadIdx.x >> 3;
        const int sub = threadIdx.x & 7;
        const int c0  = sub * 16;
        const int gr  = row0 + r;
        const float beta = 1.f / (1.f + __expf(-skip[0]));
        float el[16];
        float part = 0.f;
#pragma unroll
        for (int i = 0; i < 16; ++i) {
            const float g   = sOut[r * 128 + c0 + i];
            const float val = beta * g + (1.f - beta) * h[(size_t)gr * HH + c0 + i];
            const float e   = val > 0.f ? val : (__expf(val) - 1.f);   // ELU
            el[i] = e;
            part += e;
        }
#pragma unroll
        for (int m = 4; m >= 1; m >>= 1) part += __shfl_xor(part, m, 8);
        const float mu = part * (1.f / 128.f);
        float v2 = 0.f;
#pragma unroll
        for (int i = 0; i < 16; ++i) { const float d = el[i] - mu; v2 += d * d; }
#pragma unroll
        for (int m = 4; m >= 1; m >>= 1) v2 += __shfl_xor(v2, m, 8);
        const float inv = rsqrtf(v2 * (1.f / 128.f) + 1e-5f);
#pragma unroll
        for (int i = 0; i < 16; ++i)
            h[(size_t)gr * HH + c0 + i] =
                (el[i] - mu) * inv * ng[c0 + i] + nb[c0 + i];
    }
}

// ---------------------------------------------------------------------------
// Pooling + prediction heads
// ---------------------------------------------------------------------------
__global__ __launch_bounds__(256) void count_kernel(const int* batch, float* cnt, int n) {
    const int i = blockIdx.x * 256 + threadIdx.x;
    if (i < n) atomicAdd(&cnt[batch[i]], 1.f);
}

__global__ __launch_bounds__(256) void poolsum_kernel(
    const int* batch, const float* h, float* pooled, int n) {
    const int t = blockIdx.x * 256 + threadIdx.x;
    if (t < n * HH) {
        const int nn = t >> 7, c = t & 127;
        atomicAdd(&pooled[(size_t)batch[nn] * HH + c], h[t]);
    }
}

__global__ __launch_bounds__(256) void head_kernel(
    const float* pooled, const float* cnt,
    const float* memW, const float* memb,
    const float* timeW, const float* timeb, float* out) {
    const int b = blockIdx.x * 256 + threadIdx.x;
    if (b >= BB) return;
    float c = cnt[b];
    c = c > 1.f ? c : 1.f;
    const float inv = 1.f / c;
    float am = 0.f, at = 0.f;
#pragma unroll 4
    for (int i = 0; i < HH; ++i) {
        const float p = pooled[(size_t)b * HH + i] * inv;
        am += p * memW[i];
        at += p * timeW[i];
    }
    out[b]      = am + memb[0];
    out[BB + b] = at + timeb[0];
}

// ---------------------------------------------------------------------------
// Host orchestration
// ---------------------------------------------------------------------------
extern "C" void kernel_launch(void* const* d_in, const int* in_sizes, int n_in,
                              void* d_out, int out_size, void* d_ws, size_t ws_size,
                              hipStream_t stream) {
    (void)in_sizes; (void)n_in; (void)out_size; (void)ws_size;
    const float* x        = (const float*)d_in[0];
    const int*   ei       = (const int*)d_in[1];
    const int*   src      = ei;
    const int*   dst      = ei + EE;
    const int*   batch    = (const int*)d_in[2];
    const float* lin_op_W = (const float*)d_in[3];
    const float* lin_op_b = (const float*)d_in[4];
    const float* kW       = (const float*)d_in[5];
    const float* kb       = (const float*)d_in[6];
    const float* qW       = (const float*)d_in[7];
    const float* qb       = (const float*)d_in[8];
    const float* vW       = (const float*)d_in[9];
    const float* vb       = (const float*)d_in[10];
    const float* aW       = (const float*)d_in[11];
    const float* ab       = (const float*)d_in[12];
    const float* skip     = (const float*)d_in[13];
    const float* a_rel    = (const float*)d_in[14];
    const float* m_rel    = (const float*)d_in[15];
    const float* p_rel    = (const float*)d_in[16];
    const float* ng       = (const float*)d_in[17];
    const float* nb       = (const float*)d_in[18];
    const float* memW     = (const float*)d_in[19];
    const float* memb     = (const float*)d_in[20];
    const float* timeW    = (const float*)d_in[21];
    const float* timeb    = (const float*)d_in[22];
    float* out = (float*)d_out;

    char* ws = (char*)d_ws;
    size_t off = 0;
    auto take = [&](size_t bytes) -> void* {
        void* p = ws + off;
        off = (off + bytes + 255) & ~(size_t)255;
        return p;
    };
    __bf16* bWop = (__bf16*)take(64 * HH * sizeof(__bf16));
    __bf16* bWq  = (__bf16*)take(HH * HH * sizeof(__bf16));
    __bf16* bWk  = (__bf16*)take(HH * HH * sizeof(__bf16));
    __bf16* bWv  = (__bf16*)take(HH * HH * sizeof(__bf16));
    __bf16* bWa  = (__bf16*)take(HH * HH * sizeof(__bf16));
    float*  fbk  = (float*)take(HH * sizeof(float));
    float*  fbv  = (float*)take(HH * sizeof(float));
    float*  h    = (float*)take((size_t)NN * HH * sizeof(float));
    float*  q    = (float*)take((size_t)NN * HH * sizeof(float));
    float*  kr   = (float*)take((size_t)NN * HH * sizeof(float));
    float*  vr   = (float*)take((size_t)NN * HH * sizeof(float));
    float*  agg  = (float*)take((size_t)NN * HH * sizeof(float));
    float*  elog = (float*)take((size_t)EE * 4 * sizeof(float));
    unsigned* nmax = (unsigned*)take((size_t)NN * 4 * sizeof(unsigned));
    float*  nsum = (float*)take((size_t)NN * 4 * sizeof(float));
    float*  pooled = (float*)take((size_t)BB * HH * sizeof(float));
    float*  cnt  = (float*)take((size_t)BB * sizeof(float));

    const int rowBlocks  = NN / 16;                        // 3125 exact
    const int edgeBlocks = (EE * 4 + 255) / 256;           // 12500

    prep_kernel<<<128, 128, 0, stream>>>(lin_op_W, kW, qW, vW, aW, kb, vb,
                                         a_rel, m_rel, bWop, bWq, bWk, bWv, bWa,
                                         fbk, fbv);
    inproj_kernel<<<rowBlocks, 256, 0, stream>>>(x, bWop, lin_op_b, h);

    for (int l = 0; l < NL; ++l) {
        hipMemsetAsync(nmax, 0, (size_t)NN * 4 * sizeof(unsigned), stream);
        hipMemsetAsync(nsum, 0, (size_t)NN * 4 * sizeof(float), stream);
        hipMemsetAsync(agg,  0, (size_t)NN * HH * sizeof(float), stream);
        qkv_kernel<<<rowBlocks, 256, 0, stream>>>(h, bWq, bWk, bWv, qb, fbk, fbv,
                                                  q, kr, vr);
        edge_logits_kernel<<<edgeBlocks, 256, 0, stream>>>(src, dst, q, kr, p_rel,
                                                           elog, nmax);
        edge_expsum_kernel<<<edgeBlocks, 256, 0, stream>>>(dst, elog, nmax, nsum);
        edge_msg_kernel<<<edgeBlocks, 256, 0, stream>>>(src, dst, elog, nsum, vr, agg);
        outproj_norm_kernel<<<rowBlocks, 256, 0, stream>>>(agg, h, bWa, ab, skip,
                                                           ng, nb);
    }

    hipMemsetAsync(pooled, 0, (size_t)BB * HH * sizeof(float), stream);
    hipMemsetAsync(cnt,    0, (size_t)BB * sizeof(float), stream);
    count_kernel<<<(NN + 255) / 256, 256, 0, stream>>>(batch, cnt, NN);
    poolsum_kernel<<<((size_t)NN * HH + 255) / 256, 256, 0, stream>>>(batch, h, pooled, NN);
    head_kernel<<<(BB + 255) / 256, 256, 0, stream>>>(pooled, cnt, memW, memb,
                                                      timeW, timeb, out);
}
